// PET_18193481466478
// MI455X (gfx1250) — compile-verified, matrix operand order checked by reading, not run on hardware
//
#include <hip/hip_runtime.h>
#include <math.h>

typedef __attribute__((ext_vector_type(16))) _Float16 v16h;
typedef __attribute__((ext_vector_type(8)))  _Float16 v8h;
typedef __attribute__((ext_vector_type(4)))  _Float16 v4h;
typedef __attribute__((ext_vector_type(8)))  float    v8f;

union V16 { v16h v; v8h h[2]; };

#define B_   1024
#define P_   150
#define F_   13
#define D_   32
#define K_   10

__device__ __forceinline__ float gelu_f(float x) {
  return 0.5f * x * (1.0f + erff(x * 0.70710678118654752f));
}

// ---------------- weight pre-pack: Wt[n][k] = f16(W[k][n]), k zero-padded to Kpad ----
__global__ void k_prepw(const float* __restrict__ W, _Float16* __restrict__ Wt,
                        int Kd, int N, int Kpad)
{
  int i = blockIdx.x * blockDim.x + threadIdx.x;
  if (i >= N * Kpad) return;
  int n = i / Kpad, k = i - n * Kpad;
  Wt[i] = (k < Kd) ? (_Float16)W[(size_t)k * N + n] : (_Float16)0.0f;
}

// ---------------- generic WMMA GEMM: C = act(A(MxKd) @ W + bias) ----------------
// Wt is pre-packed f16 (N x Kpad, transposed). block = 32*(N/16) threads,
// one 16x16 tile per wave, grid.x = M/16 (M must be a multiple of 16).
__global__ void k_gemm_wmma(const float* __restrict__ A,
                            const _Float16* __restrict__ Wt,
                            const float* __restrict__ bias, float* __restrict__ C,
                            int M, int N, int Kd, int Kpad, int act)
{
  __shared__ __align__(16) _Float16 As[16 * 64];
  const int nthr = blockDim.x;
  const int m0 = blockIdx.x * 16;
  const float* At = A + (size_t)m0 * Kd;   // 16 rows, contiguous

  if ((Kd & 3) == 0) {
    const int total4 = (16 * Kd) >> 2;
    const float4* At4 = (const float4*)At;
    for (int i = threadIdx.x; i < total4; i += nthr) {
      int e0 = i << 2;
      int row = e0 / Kd, k = e0 - row * Kd;
      float4 v = At4[i];
      v4h h = { (_Float16)v.x, (_Float16)v.y, (_Float16)v.z, (_Float16)v.w };
      *(v4h*)(As + row * Kpad + k) = h;
    }
  } else {
    const int total = 16 * Kd;
    for (int i = threadIdx.x; i < total; i += nthr) {
      int row = i / Kd, k = i - row * Kd;
      As[row * Kpad + k] = (_Float16)At[i];
    }
  }
  if (Kd < Kpad) {
    const int pad = Kpad - Kd;
    for (int i = threadIdx.x; i < 16 * pad; i += nthr) {
      int row = i / pad, k = Kd + (i - row * pad);
      As[row * Kpad + k] = (_Float16)0.0f;
    }
  }
  __syncthreads();

  const int lane = threadIdx.x & 31;
  const int wave = threadIdx.x >> 5;
  const int n0   = wave * 16;
  const int half = lane >> 4;
  const int mr   = lane & 15;

  const v8h* Ar8 = (const v8h*)(As + mr * Kpad);
  const v8h* Bt8 = (const v8h*)(Wt + (size_t)(n0 + mr) * Kpad);

  v8f acc = {};
  for (int k0 = 0; k0 < Kpad; k0 += 32) {
    int i0 = (k0 >> 3) + half;
    V16 a, b;
    a.h[0] = Ar8[i0];  a.h[1] = Ar8[i0 + 2];
    b.h[0] = Bt8[i0];  b.h[1] = Bt8[i0 + 2];
    acc = __builtin_amdgcn_wmma_f32_16x16x32_f16(false, a.v, false, b.v, (short)0, acc,
                                                 false, false);
  }
#pragma unroll
  for (int v = 0; v < 8; ++v) {
    int m = m0 + v + 8 * half;
    int n = n0 + mr;
    float val = acc[v] + bias[n];
    if (act == 1) val = gelu_f(val);
    C[(size_t)m * N + n] = val;
  }
}

// ---------------- mask & points ----------------
__global__ void k_mask(const float* __restrict__ x, float* __restrict__ mask, int n)
{
  int i = blockIdx.x * blockDim.x + threadIdx.x;
  if (i < n) mask[i] = (x[(size_t)i * F_] != 0.0f) ? 1.0f : 0.0f;
}

__global__ void k_make_points(const float* __restrict__ src, int stride, int dim,
                              const float* __restrict__ mask, float* __restrict__ pts,
                              int rows)
{
  int i = blockIdx.x * blockDim.x + threadIdx.x;
  if (i >= rows * dim) return;
  int r = i / dim, c = i - r * dim;
  pts[i] = src[(size_t)r * stride + c] + 999.0f * (1.0f - mask[r]);
}

// ---------------- kNN: block per batch, thread per point ----------------
template <int DIM>
__global__ void k_knn(const float* __restrict__ pts, int* __restrict__ idx)
{
  __shared__ float sp[P_ * DIM];
  const int b = blockIdx.x;
  for (int t = threadIdx.x; t < P_ * DIM; t += blockDim.x)
    sp[t] = pts[(size_t)b * P_ * DIM + t];
  __syncthreads();
  const int p = threadIdx.x;
  if (p >= P_) return;
  float mine[DIM];
#pragma unroll
  for (int c = 0; c < DIM; ++c) mine[c] = sp[p * DIM + c];
  float bd[K_]; int bi[K_];
#pragma unroll
  for (int j = 0; j < K_; ++j) { bd[j] = 3.0e38f; bi[j] = 0; }
  for (int q = 0; q < P_; ++q) {
    if (q == p) continue;
    float d = 0.0f;
#pragma unroll
    for (int c = 0; c < DIM; ++c) {
      float t = mine[c] - sp[q * DIM + c];
      d += t * t;
    }
    if (d < bd[K_ - 1]) {
      int j = K_ - 1;
      while (j > 0 && bd[j - 1] > d) { bd[j] = bd[j - 1]; bi[j] = bi[j - 1]; --j; }
      bd[j] = d; bi[j] = q;
    }
  }
  int* op = idx + ((size_t)b * P_ + p) * K_;
  for (int j = 0; j < K_; ++j) op[j] = bi[j];
}

// ---------------- fused EdgeConv: block (128 thr) per (b,p) ----------------
// E(16 x Cin f16, zero-padded) -> H = gelu(E@w1+b1) (16x64 f16) -> O = gelu(H@w2+b2)
// (16x32 f32) -> mean over K rows. w1t/w2t pre-packed f16 transposed.
__global__ void k_edgeconv(const float* __restrict__ feats, int fs, int Fin,
                           const int* __restrict__ idx,
                           const _Float16* __restrict__ w1t, const float* __restrict__ b1,
                           const _Float16* __restrict__ w2t, const float* __restrict__ b2,
                           float* __restrict__ out)
{
  const int Cin   = 2 * Fin;                // 26 or 64
  const int Kpad1 = (Cin + 31) & ~31;       // 32 or 64
  const int bp  = blockIdx.x;
  const int b   = bp / P_;
  __shared__ __align__(16) _Float16 Ef[16 * 64];
  __shared__ __align__(16) _Float16 Hf[16 * 64];
  __shared__ float O[16 * 32];
  const int tid = threadIdx.x;
  const float* fc = feats + (size_t)bp * fs;

  for (int t = tid; t < 16 * Kpad1; t += blockDim.x) {
    int kk = t / Kpad1, c = t - kk * Kpad1;
    float v = 0.0f;
    if (kk < K_ && c < Cin) {
      if (c < Fin) {
        int nb = idx[(size_t)bp * K_ + kk];
        v = feats[((size_t)b * P_ + nb) * fs + c] - fc[c];
      } else {
        v = fc[c - Fin];
      }
    }
    Ef[t] = (_Float16)v;
  }
  __syncthreads();

  const int lane = tid & 31, wave = tid >> 5;
  const int half = lane >> 4, mr = lane & 15;

  { // GEMM1: all 4 waves, N=64
    const int n0 = wave * 16;
    const v8h* Ar8 = (const v8h*)(Ef + mr * Kpad1);
    const v8h* Bt8 = (const v8h*)(w1t + (size_t)(n0 + mr) * Kpad1);
    v8f acc = {};
    for (int k0 = 0; k0 < Kpad1; k0 += 32) {
      int i0 = (k0 >> 3) + half;
      V16 a, bb;
      a.h[0]  = Ar8[i0];  a.h[1]  = Ar8[i0 + 2];
      bb.h[0] = Bt8[i0];  bb.h[1] = Bt8[i0 + 2];
      acc = __builtin_amdgcn_wmma_f32_16x16x32_f16(false, a.v, false, bb.v, (short)0,
                                                   acc, false, false);
    }
#pragma unroll
    for (int v = 0; v < 8; ++v) {
      int m = v + 8 * half, n = n0 + mr;
      Hf[m * 64 + n] = (_Float16)gelu_f(acc[v] + b1[n]);
    }
  }
  __syncthreads();

  if (wave < 2) { // GEMM2: N=32, K=64
    const int n0 = wave * 16;
    const v8h* Ar8 = (const v8h*)(Hf + mr * 64);
    const v8h* Bt8 = (const v8h*)(w2t + (size_t)(n0 + mr) * 64);
    v8f acc = {};
#pragma unroll
    for (int k0 = 0; k0 < 64; k0 += 32) {
      int i0 = (k0 >> 3) + half;
      V16 a, bb;
      a.h[0]  = Ar8[i0];  a.h[1]  = Ar8[i0 + 2];
      bb.h[0] = Bt8[i0];  bb.h[1] = Bt8[i0 + 2];
      acc = __builtin_amdgcn_wmma_f32_16x16x32_f16(false, a.v, false, bb.v, (short)0,
                                                   acc, false, false);
    }
#pragma unroll
    for (int v = 0; v < 8; ++v) {
      int m = v + 8 * half, n = n0 + mr;
      O[m * 32 + n] = gelu_f(acc[v] + b2[n]);
    }
  }
  __syncthreads();

  if (tid < 32) {
    float s = 0.0f;
#pragma unroll
    for (int r = 0; r < K_; ++r) s += O[r * 32 + tid];
    out[(size_t)bp * 32 + tid] = s * (1.0f / (float)K_);
  }
}

// ---------------- GroupNorm (single group over T*D per batch) ----------------
__global__ void k_gn(const float* __restrict__ src, float* __restrict__ dst,
                     const float* __restrict__ g, const float* __restrict__ be, int T)
{
  const int b = blockIdx.x;
  const int n = T * D_;
  const float* xb = src + (size_t)b * n;
  float s = 0.f, s2 = 0.f;
  for (int i = threadIdx.x; i < n; i += blockDim.x) {
    float v = xb[i]; s += v; s2 += v * v;
  }
  __shared__ float r1[256], r2[256];
  r1[threadIdx.x] = s; r2[threadIdx.x] = s2;
  __syncthreads();
  for (int st = 128; st > 0; st >>= 1) {
    if (threadIdx.x < st) {
      r1[threadIdx.x] += r1[threadIdx.x + st];
      r2[threadIdx.x] += r2[threadIdx.x + st];
    }
    __syncthreads();
  }
  float mu  = r1[0] / (float)n;
  float var = r2[0] / (float)n - mu * mu;
  float inv = rsqrtf(var + 0.001f);
  float* db = dst + (size_t)b * n;
  for (int i = threadIdx.x; i < n; i += blockDim.x) {
    int c = i & (D_ - 1);
    db[i] = (xb[i] - mu) * inv * g[c] + be[c];
  }
}

// ---------------- attention: block per (b,h), K/V in LDS, 2-pass softmax ----------------
template <int KD>
__global__ void k_attn(const float* __restrict__ q, const float* __restrict__ kmat,
                       const float* __restrict__ vmat, float* __restrict__ out,
                       int rowsQ, int S, int Tq, int H, float scale)
{
  const int b = blockIdx.x / H;
  const int h = blockIdx.x % H;
  const int HKD = H * KD;
  __shared__ __align__(16) float KS[151 * 16];
  __shared__ __align__(16) float VS[151 * 16];
  const float* kp = kmat + (size_t)b * S * HKD + h * KD;
  const float* vp = vmat + (size_t)b * S * HKD + h * KD;
  for (int t = threadIdx.x; t < S * KD; t += blockDim.x) {
    int s = t / KD, j = t - s * KD;
    KS[t] = kp[(size_t)s * HKD + j];
    VS[t] = vp[(size_t)s * HKD + j];
  }
  __syncthreads();
  const int t = threadIdx.x;
  if (t >= Tq) return;
  const float* qr = q + ((size_t)b * rowsQ + t) * HKD + h * KD;
  float qv[KD];
#pragma unroll
  for (int j = 0; j < KD; ++j) qv[j] = qr[j] * scale;
  float mx = -3.0e38f;
  for (int s = 0; s < S; ++s) {
    const float4* kr = (const float4*)(KS + s * KD);
    float d = 0.f;
#pragma unroll
    for (int r = 0; r < KD / 4; ++r) {
      float4 kk = kr[r];
      d += qv[4*r] * kk.x + qv[4*r+1] * kk.y + qv[4*r+2] * kk.z + qv[4*r+3] * kk.w;
    }
    mx = fmaxf(mx, d);
  }
  float sum = 0.f;
  float o[KD];
#pragma unroll
  for (int j = 0; j < KD; ++j) o[j] = 0.f;
  for (int s = 0; s < S; ++s) {
    const float4* kr = (const float4*)(KS + s * KD);
    const float4* vr = (const float4*)(VS + s * KD);
    float d = 0.f;
#pragma unroll
    for (int r = 0; r < KD / 4; ++r) {
      float4 kk = kr[r];
      d += qv[4*r] * kk.x + qv[4*r+1] * kk.y + qv[4*r+2] * kk.z + qv[4*r+3] * kk.w;
    }
    float e = __expf(d - mx);
    sum += e;
#pragma unroll
    for (int r = 0; r < KD / 4; ++r) {
      float4 vv = vr[r];
      o[4*r] += e * vv.x; o[4*r+1] += e * vv.y; o[4*r+2] += e * vv.z; o[4*r+3] += e * vv.w;
    }
  }
  float inv = 1.f / sum;
  float* orow = out + ((size_t)b * Tq + t) * HKD + h * KD;
#pragma unroll
  for (int j = 0; j < KD; ++j) orow[j] = o[j] * inv;
}

// ---------------- elementwise helpers ----------------
__global__ void k_add2(float* __restrict__ out, const float* __restrict__ a,
                       const float* __restrict__ b2, int n)
{
  int i = blockIdx.x * blockDim.x + threadIdx.x;
  if (i < n) out[i] = a[i] + b2[i];
}

__global__ void k_rowmask(float* __restrict__ out, const float* __restrict__ a,
                          const float* __restrict__ mask, int n)
{
  int i = blockIdx.x * blockDim.x + threadIdx.x;
  if (i < n) out[i] = a[i] * mask[i >> 5];
}

__global__ void k_ls_add(float* __restrict__ out, const float* __restrict__ a,
                         const float* __restrict__ addv, const float* __restrict__ ls,
                         const float* __restrict__ mask, int n, int use_mask,
                         int post_mask)
{
  int i = blockIdx.x * blockDim.x + threadIdx.x;
  if (i >= n) return;
  float m = 1.0f;
  if (use_mask) m = mask[i >> 5];
  float v = a[i] * ls[i & 31] * m + addv[i];
  if (post_mask) v *= m;
  out[i] = v;
}

__global__ void k_cls_init(const float* __restrict__ clsP, float* __restrict__ cls)
{
  int i = blockIdx.x * blockDim.x + threadIdx.x;
  if (i < B_ * D_) cls[i] = clsP[i & 31];
}

__global__ void k_concat(const float* __restrict__ cls, const float* __restrict__ body,
                         float* __restrict__ cat)
{
  int i = blockIdx.x * blockDim.x + threadIdx.x;
  const int n = B_ * (P_ + 1) * D_;
  if (i >= n) return;
  int c = i & 31;
  int r = i >> 5;
  int b = r / (P_ + 1);
  int t = r - b * (P_ + 1);
  cat[i] = (t == 0) ? cls[(size_t)b * D_ + c]
                    : body[((size_t)b * P_ + (t - 1)) * D_ + c];
}

__global__ void k_out(const float* __restrict__ cls, const float* __restrict__ w,
                      const float* __restrict__ bt, float* __restrict__ out)
{
  int b = blockIdx.x * blockDim.x + threadIdx.x;
  if (b >= B_) return;
  float s = bt[0];
#pragma unroll
  for (int c = 0; c < D_; ++c) s += cls[(size_t)b * D_ + c] * w[c];
  out[b] = s;
}

// ---------------- host side ----------------
static inline void gemm(const float* A, const _Float16* Wt, const float* bias, float* C,
                        int M, int N, int Kd, int act, hipStream_t s)
{
  int Kpad = (Kd + 31) & ~31;
  k_gemm_wmma<<<dim3(M / 16), dim3(32 * (N / 16)), 0, s>>>(A, Wt, bias, C, M, N, Kd,
                                                           Kpad, act);
}

extern "C" void kernel_launch(void* const* d_in, const int* in_sizes, int n_in,
                              void* d_out, int out_size, void* d_ws, size_t ws_size,
                              hipStream_t stream)
{
  (void)in_sizes; (void)n_in; (void)out_size; (void)ws_size;
  const float* x = (const float*)d_in[0];
  auto PRM = [&](int i) { return (const float*)d_in[i]; };

  const int BP   = B_ * P_;
  const int BPD  = BP * D_;
  const int CATR = B_ * (P_ + 1);
  const int CATD = CATR * D_;

  char* base = (char*)d_ws;
  size_t off = 0;
  auto allocB = [&](size_t bytes) -> void* {
    void* p = base + off;
    off = (off + bytes + 255) & ~(size_t)255;
    return p;
  };
  auto allocF = [&](size_t elems) -> float* { return (float*)allocB(elems * 4); };

  float* mask = allocF(BP);
  float* h64  = allocF((size_t)BP * 64);   // enc hidden / FFN hidden / fB (reused)
  float* enc  = allocF(BPD);
  float* fA   = allocF(BPD);               // local feats / body (reused)
  float* x1   = allocF(CATD);              // gn output / points (reused)
  float* qb   = allocF(CATD);
  float* kb   = allocF(CATD);
  float* vb   = allocF(CATD);
  float* ao   = allocF(BPD);
  float* skip = allocF(BPD);
  float* x2   = allocF(BPD);
  float* x3   = allocF(BPD);
  float* cat  = allocF(CATD);
  float* cls  = allocF(B_ * D_);
  float* updg = allocF(B_ * D_);
  float* x2h  = allocF(B_ * D_);
  float* x3h  = allocF(B_ * D_);
  float* x3g  = allocF(B_ * D_);
  float* clsg = allocF(B_ * D_);
  int*   idx  = (int*)allocB((size_t)BP * K_ * 4);
  float* fB   = h64;                       // alias: free during local stage
  float* pts  = x1;                        // alias: free during local stage

  // ---- pre-pack all GEMM weights to transposed padded f16 ----
  auto prep = [&](int prmIdx, int Kd, int N) -> const _Float16* {
    int Kpad = (Kd + 31) & ~31;
    _Float16* dst = (_Float16*)allocB((size_t)N * Kpad * 2);
    int n = N * Kpad;
    k_prepw<<<dim3((n + 255) / 256), 256, 0, stream>>>(PRM(prmIdx), dst, Kd, N, Kpad);
    return dst;
  };
  const _Float16* encW1t = prep(1, F_, 64);
  const _Float16* encW2t = prep(3, 64, D_);
  const _Float16* lw1t[2] = { prep(5, 2 * F_, 64), prep(9, 64, 64) };
  const _Float16* lw2t[2] = { prep(7, 64, D_),     prep(11, 64, D_) };
  struct BW { const _Float16 *wq, *wk, *wv, *wo, *f1, *f2; };
  BW trw[2], hdw[2];
  for (int t = 0; t < 2; ++t) {
    int pb = 13 + t * 18;
    trw[t].wq = prep(pb + 2,  D_, D_);
    trw[t].wk = prep(pb + 4,  D_, D_);
    trw[t].wv = prep(pb + 6,  D_, D_);
    trw[t].wo = prep(pb + 8,  D_, D_);
    trw[t].f1 = prep(pb + 13, D_, 64);
    trw[t].f2 = prep(pb + 15, 64, D_);
  }
  for (int h = 0; h < 2; ++h) {
    int pb = 49 + h * 20;
    hdw[h].wq = prep(pb + 2,  D_, D_);
    hdw[h].wk = prep(pb + 4,  D_, D_);
    hdw[h].wv = prep(pb + 6,  D_, D_);
    hdw[h].wo = prep(pb + 8,  D_, D_);
    hdw[h].f1 = prep(pb + 15, D_, 64);
    hdw[h].f2 = prep(pb + 17, 64, D_);
  }

  auto ew = [](int n) { return dim3((n + 255) / 256); };

  // mask + encoder MLP
  k_mask<<<ew(BP), 256, 0, stream>>>(x, mask, BP);
  gemm(x, encW1t, PRM(2), h64, BP, 64, F_, 1, stream);
  gemm(h64, encW2t, PRM(4), enc, BP, D_, 64, 1, stream);

  // local block 1 (points = x[:,:,:2] + shift, feats = x)
  k_make_points<<<ew(BP * 2), 256, 0, stream>>>(x, F_, 2, mask, pts, BP);
  k_knn<2><<<B_, 160, 0, stream>>>(pts, idx);
  k_edgeconv<<<BP, 128, 0, stream>>>(x, F_, F_, idx, lw1t[0], PRM(6), lw2t[0], PRM(8),
                                     fA);
  // local block 2 (points = feats = fA)
  k_make_points<<<ew(BPD), 256, 0, stream>>>(fA, D_, D_, mask, pts, BP);
  k_knn<D_><<<B_, 160, 0, stream>>>(pts, idx);
  k_edgeconv<<<BP, 128, 0, stream>>>(fA, D_, D_, idx, lw1t[1], PRM(10), lw2t[1],
                                     PRM(12), fB);

  k_add2<<<ew(BPD), 256, 0, stream>>>(enc, fB, enc, BPD);
  k_rowmask<<<ew(BPD), 256, 0, stream>>>(skip, enc, mask, BPD);

  // transformer blocks
  for (int t = 0; t < 2; ++t) {
    int pb = 13 + t * 18;
    k_gn<<<B_, 256, 0, stream>>>(enc, x1, PRM(pb + 0), PRM(pb + 1), P_);
    gemm(x1, trw[t].wq, PRM(pb + 3), qb, BP, D_, D_, 0, stream);
    gemm(x1, trw[t].wk, PRM(pb + 5), kb, BP, D_, D_, 0, stream);
    gemm(x1, trw[t].wv, PRM(pb + 7), vb, BP, D_, D_, 0, stream);
    k_attn<16><<<B_ * 2, 160, 0, stream>>>(qb, kb, vb, ao, P_, P_, P_, 2, 0.25f);
    gemm(ao, trw[t].wo, PRM(pb + 9), x3, BP, D_, D_, 0, stream);
    k_ls_add<<<ew(BPD), 256, 0, stream>>>(x2, x3, enc, PRM(pb + 10), mask, BPD, 1, 0);
    k_gn<<<B_, 256, 0, stream>>>(x2, x1, PRM(pb + 11), PRM(pb + 12), P_);
    gemm(x1, trw[t].f1, PRM(pb + 14), h64, BP, 64, D_, 1, stream);
    gemm(h64, trw[t].f2, PRM(pb + 16), x3, BP, D_, 64, 0, stream);
    k_ls_add<<<ew(BPD), 256, 0, stream>>>(enc, x3, x2, PRM(pb + 17), mask, BPD, 1, 1);
  }

  float* body = fA;
  k_add2<<<ew(BPD), 256, 0, stream>>>(body, enc, skip, BPD);
  k_cls_init<<<ew(B_ * D_), 256, 0, stream>>>(PRM(89), cls);

  // class-attention head blocks
  for (int hh = 0; hh < 2; ++hh) {
    int pb = 49 + hh * 20;
    k_concat<<<ew(CATD), 256, 0, stream>>>(cls, body, cat);
    k_gn<<<B_, 256, 0, stream>>>(cat, x1, PRM(pb + 0), PRM(pb + 1), P_ + 1);
    gemm(x1, hdw[hh].wq, PRM(pb + 3), qb, CATR, D_, D_, 0, stream);
    gemm(x1, hdw[hh].wk, PRM(pb + 5), kb, CATR, D_, D_, 0, stream);
    gemm(x1, hdw[hh].wv, PRM(pb + 7), vb, CATR, D_, D_, 0, stream);
    k_attn<8><<<B_ * 4, 192, 0, stream>>>(qb, kb, vb, ao, P_ + 1, P_ + 1, 1, 4,
                                          0.35355339059327373f);
    gemm(ao, hdw[hh].wo, PRM(pb + 9), x3h, B_, D_, D_, 0, stream);
    k_gn<<<B_, 256, 0, stream>>>(x3h, updg, PRM(pb + 10), PRM(pb + 11), 1);
    k_ls_add<<<ew(B_ * D_), 256, 0, stream>>>(x2h, updg, cls, PRM(pb + 12), mask,
                                              B_ * D_, 0, 0);
    k_gn<<<B_, 256, 0, stream>>>(x2h, x3g, PRM(pb + 13), PRM(pb + 14), 1);
    gemm(x3g, hdw[hh].f1, PRM(pb + 16), h64, B_, 64, D_, 1, stream);
    gemm(h64, hdw[hh].f2, PRM(pb + 18), x3h, B_, D_, 64, 0, stream);
    k_ls_add<<<ew(B_ * D_), 256, 0, stream>>>(cls, x3h, x2h, PRM(pb + 19), mask,
                                              B_ * D_, 0, 0);
  }

  k_gn<<<B_, 256, 0, stream>>>(cls, clsg, PRM(90), PRM(91), 1);
  k_out<<<ew(B_), 256, 0, stream>>>(clsg, PRM(92), PRM(93), (float*)d_out);
}
